// multiheadkvcache_76141180223901
// MI455X (gfx1250) — compile-verified, matrix operand order checked by reading, not run on hardware
//
#include <hip/hip_runtime.h>
#include <hip/hip_bf16.h>
#include <stdint.h>
#include <stddef.h>

// ---------------------------------------------------------------------------
// Problem constants: B=4, N=2048, DIN=DOUT=2048, H=16, DH=128
// ---------------------------------------------------------------------------
#define BATCH   4
#define SEQ     2048
#define DMODEL  2048
#define NHEAD   16
#define DHEAD   128
#define MROWS   (BATCH * SEQ)          // 8192 rows in the big GEMMs

typedef __attribute__((ext_vector_type(16))) __bf16 v16bf;
typedef __attribute__((ext_vector_type(8)))  float  v8f;

__device__ __forceinline__ v8f vzero8() {
    v8f z;
#pragma unroll
    for (int i = 0; i < 8; ++i) z[i] = 0.0f;
    return z;
}

__device__ __forceinline__ v8f wmma_bf16(v16bf a, v16bf b, v8f c) {
    // D = A(16x32 bf16) * B(32x16 bf16) + C(16x16 f32)
    return __builtin_amdgcn_wmma_f32_16x16x32_bf16(
        /*neg_a=*/false, a, /*neg_b=*/false, b,
        /*c_mod=*/(short)0, c, /*reuse_a=*/false, /*reuse_b=*/false);
}

// ---------------------------------------------------------------------------
// 1) fp32 -> bf16 elementwise convert
// ---------------------------------------------------------------------------
__global__ void cvt_bf16_kernel(const float* __restrict__ in,
                                __bf16* __restrict__ out, int n) {
    int i = blockIdx.x * blockDim.x + threadIdx.x;
    if (i < n) out[i] = (__bf16)in[i];
}

// ---------------------------------------------------------------------------
// 2) Wt[n][k] = (bf16) W[k][n]   (2048 x 2048, LDS-tiled transpose)
// ---------------------------------------------------------------------------
__global__ __launch_bounds__(256)
void transpose_cvt_kernel(const float* __restrict__ W, __bf16* __restrict__ Wt) {
    __shared__ float tile[32][33];
    const int bx = blockIdx.x * 32;   // n block
    const int by = blockIdx.y * 32;   // k block
    const int tx = threadIdx.x;       // 0..31
    const int ty = threadIdx.y;       // 0..7
#pragma unroll
    for (int i = 0; i < 32; i += 8)
        tile[ty + i][tx] = W[(size_t)(by + ty + i) * DMODEL + bx + tx];
    __syncthreads();
#pragma unroll
    for (int i = 0; i < 32; i += 8)
        Wt[(size_t)(bx + ty + i) * DMODEL + by + tx] = (__bf16)tile[tx][ty + i];
}

// ---------------------------------------------------------------------------
// 3) GEMM: C[M,N] = A[M,K](bf16) * Bt[N,K]^T(bf16) + bias
//    One wave computes a 16x128 tile (8 accumulators), software-pipelined:
//    ping/pong fragment buffers so loads for step k+32 are in flight while
//    WMMAs consume step k.
//    mode 0: fp32 row-major [M,N]; mode 1: bf16 [B,H,N,DH]; mode 2: bf16 [B,H,DH,N]
// ---------------------------------------------------------------------------
__global__ __launch_bounds__(128)
void gemm_bias_kernel(const __bf16* __restrict__ A, const __bf16* __restrict__ Bt,
                      const float* __restrict__ bias,
                      float* __restrict__ Cf, __bf16* __restrict__ Cb,
                      int M, int N, int K, int mode) {
    const int lane = threadIdx.x & 31;
    const int wid  = threadIdx.x >> 5;
    const int ntn  = N >> 7;                       // 128-wide column tiles
    const int tile = blockIdx.x * 4 + wid;
    const int tm   = tile / ntn;
    const int tn   = tile % ntn;
    if (tm * 16 >= M) return;

    const int m = lane & 15;
    const int h = lane >> 4;

    v8f acc[8];
#pragma unroll
    for (int j = 0; j < 8; ++j) acc[j] = vzero8();

    const __bf16* Ap = A  + (size_t)(tm * 16 + m) * K + 16 * h;
    const __bf16* Bp = Bt + (size_t)(tn * 128 + m) * K + 16 * h;  // + j*16*K per subtile

    v16bf aA, aB, bA[8], bB[8];
    aA = *(const v16bf*)(Ap);
#pragma unroll
    for (int j = 0; j < 8; ++j)
        bA[j] = *(const v16bf*)(Bp + (size_t)j * 16 * K);

    for (int k = 0; k < K; k += 64) {
        // prefetch the A stream one super-step ahead
        __builtin_prefetch(Ap + k + 128, 0, 1);
        // load ping buffer for k+32 while consuming pong (k)
        aB = *(const v16bf*)(Ap + k + 32);
#pragma unroll
        for (int j = 0; j < 8; ++j)
            bB[j] = *(const v16bf*)(Bp + (size_t)j * 16 * K + k + 32);
#pragma unroll
        for (int j = 0; j < 8; ++j) acc[j] = wmma_bf16(aA, bA[j], acc[j]);

        if (k + 64 < K) {
            aA = *(const v16bf*)(Ap + k + 64);
#pragma unroll
            for (int j = 0; j < 8; ++j)
                bA[j] = *(const v16bf*)(Bp + (size_t)j * 16 * K + k + 64);
        }
#pragma unroll
        for (int j = 0; j < 8; ++j) acc[j] = wmma_bf16(aB, bB[j], acc[j]);
    }

#pragma unroll
    for (int j = 0; j < 8; ++j) {
        const int col = tn * 128 + j * 16 + m;
        const float bv = bias[col];
#pragma unroll
        for (int r = 0; r < 8; ++r) {
            const float v   = acc[j][r] + bv;
            const int   row = tm * 16 + r + 8 * h;
            if (mode == 0) {
                Cf[(size_t)row * N + col] = v;
            } else {
                const int head = col >> 7;            // / DHEAD
                const int d    = col & (DHEAD - 1);
                const int b_   = row >> 11;           // / SEQ
                const int nn   = row & (SEQ - 1);
                size_t addr;
                if (mode == 1)       // [B,H,N,DH]
                    addr = ((size_t)(b_ * NHEAD + head) * SEQ + nn) * DHEAD + d;
                else                 // [B,H,DH,N]
                    addr = ((size_t)(b_ * NHEAD + head) * DHEAD + d) * SEQ + nn;
                Cb[addr] = (__bf16)v;
            }
        }
    }
}

// ---------------------------------------------------------------------------
// 4) Causal flash attention. One wave per (b, h, 16-row q tile).
//    kv step = 32. All K-fragment loads are issued before the score WMMAs;
//    all V-fragment loads are issued before the softmax VALU block so the
//    softmax arithmetic hides their latency.
// ---------------------------------------------------------------------------
__global__ __launch_bounds__(128)
void attn_kernel(const __bf16* __restrict__ Q, const __bf16* __restrict__ K,
                 const __bf16* __restrict__ Vt, __bf16* __restrict__ ctx) {
    __shared__ __bf16 Pt[4][16 * 32];           // per-wave 16x32 P tile

    const int lane = threadIdx.x & 31;
    const int wid  = threadIdx.x >> 5;
    const int gw   = blockIdx.x * 4 + wid;       // global wave id
    const int bh   = gw >> 7;                    // / (SEQ/16 = 128)
    const int qt   = gw & 127;
    const int q0   = qt * 16;
    const int m    = lane & 15;
    const int h    = lane >> 4;
    const float scale = 0.08838834764831845f;    // 1/sqrt(128)

    const __bf16* Qbh = Q  + (size_t)bh * SEQ * DHEAD;
    const __bf16* Kbh = K  + (size_t)bh * SEQ * DHEAD;
    const __bf16* Vbh = Vt + (size_t)bh * DHEAD * SEQ;

    // Preload Q fragments for this q tile (16 x 128 -> 4 chunks of K=32).
    v16bf qf[4];
#pragma unroll
    for (int c = 0; c < 4; ++c)
        qf[c] = *(const v16bf*)(Qbh + (size_t)(q0 + m) * DHEAD + c * 32 + 16 * h);

    v8f acc[8];
#pragma unroll
    for (int j = 0; j < 8; ++j) acc[j] = vzero8();
    float mrow[8], lrow[8];
#pragma unroll
    for (int r = 0; r < 8; ++r) { mrow[r] = -1e30f; lrow[r] = 0.0f; }

    __bf16* P = Pt[wid];
    const int nsteps = qt / 2 + 1;               // last kv step covers the diagonal

    for (int step = 0; step < nsteps; ++step) {
        const int k0 = step * 32;

        // ---- issue all K-fragment loads, then the 8 score WMMAs ----
        v16bf kf0[4], kf1[4];
#pragma unroll
        for (int c = 0; c < 4; ++c) {
            kf0[c] = *(const v16bf*)(Kbh + (size_t)(k0 + m)      * DHEAD + c * 32 + 16 * h);
            kf1[c] = *(const v16bf*)(Kbh + (size_t)(k0 + 16 + m) * DHEAD + c * 32 + 16 * h);
        }
        v8f s0 = vzero8(), s1 = vzero8();
#pragma unroll
        for (int c = 0; c < 4; ++c) {
            s0 = wmma_bf16(qf[c], kf0[c], s0);
            s1 = wmma_bf16(qf[c], kf1[c], s1);
        }

        // ---- issue V-fragment loads early; softmax VALU hides the latency ----
        v16bf vf[8];
#pragma unroll
        for (int j = 0; j < 8; ++j)
            vf[j] = *(const v16bf*)(Vbh + (size_t)(j * 16 + m) * SEQ + k0 + 16 * h);

        const bool last = (step == nsteps - 1);  // only the final step crosses diag
#pragma unroll
        for (int r = 0; r < 8; ++r) {
            float a0 = s0[r] * scale;
            float a1 = s1[r] * scale;
            if (last) {
                const int qg = q0 + r + 8 * h;
                if (k0 + m      > qg) a0 = -1e30f;
                if (k0 + 16 + m > qg) a1 = -1e30f;
            }
            // row max across the 16 lanes of this half
            float mx = fmaxf(a0, a1);
#pragma unroll
            for (int msk = 1; msk < 16; msk <<= 1)
                mx = fmaxf(mx, __shfl_xor(mx, msk, 32));
            const float mnew = fmaxf(mrow[r], mx);
            const float corr = __expf(mrow[r] - mnew);
            const float p0 = __expf(a0 - mnew);
            const float p1 = __expf(a1 - mnew);
            float rs = p0 + p1;
#pragma unroll
            for (int msk = 1; msk < 16; msk <<= 1)
                rs += __shfl_xor(rs, msk, 32);
            lrow[r] = lrow[r] * corr + rs;
            mrow[r] = mnew;
#pragma unroll
            for (int j = 0; j < 8; ++j) acc[j][r] *= corr;
            // stage P (C-layout -> LDS row-major 16x32)
            P[(r + 8 * h) * 32 + m]      = (__bf16)p0;
            P[(r + 8 * h) * 32 + m + 16] = (__bf16)p1;
        }

        // Re-fragment P as WMMA A operand (wave-private LDS; DS ops in-order).
        v16bf pf = *(const v16bf*)(P + m * 32 + 16 * h);
#pragma unroll
        for (int j = 0; j < 8; ++j)
            acc[j] = wmma_bf16(pf, vf[j], acc[j]);
    }

    // Normalize and write ctx as bf16 row-major [B*N, DMODEL].
    const int b_   = bh >> 4;
    const int head = bh & (NHEAD - 1);
#pragma unroll
    for (int r = 0; r < 8; ++r) {
        const float inv = 1.0f / lrow[r];
        const int nn = q0 + r + 8 * h;
        const size_t rowbase = (size_t)(b_ * SEQ + nn) * DMODEL + head * DHEAD;
#pragma unroll
        for (int j = 0; j < 8; ++j)
            ctx[rowbase + j * 16 + m] = (__bf16)(acc[j][r] * inv);
    }
}

// ---------------------------------------------------------------------------
// Launch
// ---------------------------------------------------------------------------
extern "C" void kernel_launch(void* const* d_in, const int* in_sizes, int n_in,
                              void* d_out, int out_size, void* d_ws, size_t ws_size,
                              hipStream_t stream) {
    const float* X  = (const float*)d_in[0];
    const float* Wq = (const float*)d_in[1];
    const float* bq = (const float*)d_in[2];
    const float* Wk = (const float*)d_in[3];
    const float* bk = (const float*)d_in[4];
    const float* Wv = (const float*)d_in[5];
    const float* bv = (const float*)d_in[6];
    const float* Wo = (const float*)d_in[7];
    const float* bo = (const float*)d_in[8];
    float* out = (float*)d_out;

    char* ws = (char*)d_ws;
    const size_t MB = 1ull << 20;
    __bf16* Xb   = (__bf16*)(ws);              //  32 MB  [8192, 2048]
    __bf16* Wtq  = (__bf16*)(ws +  32 * MB);   //   8 MB  [2048, 2048] (N,K)
    __bf16* Wtk  = (__bf16*)(ws +  40 * MB);
    __bf16* Wtv  = (__bf16*)(ws +  48 * MB);
    __bf16* Wto  = (__bf16*)(ws +  56 * MB);
    __bf16* Qb   = (__bf16*)(ws +  64 * MB);   //  32 MB  [B,H,N,DH]
    __bf16* Kb   = (__bf16*)(ws +  96 * MB);   //  32 MB  [B,H,N,DH]
    __bf16* Vtb  = (__bf16*)(ws + 128 * MB);   //  32 MB  [B,H,DH,N]
    __bf16* Ctx  = (__bf16*)(ws + 160 * MB);   //  32 MB  [8192, 2048]

    // 1) convert X
    {
        const int n = MROWS * DMODEL;
        cvt_bf16_kernel<<<n / 256, 256, 0, stream>>>(X, Xb, n);
    }
    // 2) transpose+convert weights
    {
        dim3 g(DMODEL / 32, DMODEL / 32), b(32, 8);
        transpose_cvt_kernel<<<g, b, 0, stream>>>(Wq, Wtq);
        transpose_cvt_kernel<<<g, b, 0, stream>>>(Wk, Wtk);
        transpose_cvt_kernel<<<g, b, 0, stream>>>(Wv, Wtv);
        transpose_cvt_kernel<<<g, b, 0, stream>>>(Wo, Wto);
    }
    // 3) projections (tiles = (M/16)*(N/128) = 512*16 = 8192; 4 waves/block)
    {
        const int blocks = (MROWS / 16) * (DMODEL / 128) / 4;
        gemm_bias_kernel<<<blocks, 128, 0, stream>>>(Xb, Wtq, bq, nullptr, Qb,
                                                     MROWS, DMODEL, DMODEL, 1);
        gemm_bias_kernel<<<blocks, 128, 0, stream>>>(Xb, Wtk, bk, nullptr, Kb,
                                                     MROWS, DMODEL, DMODEL, 1);
        gemm_bias_kernel<<<blocks, 128, 0, stream>>>(Xb, Wtv, bv, nullptr, Vtb,
                                                     MROWS, DMODEL, DMODEL, 2);
    }
    // 4) attention (waves = B*H*(N/16) = 8192; 4 waves/block)
    attn_kernel<<<8192 / 4, 128, 0, stream>>>(Qb, Kb, Vtb, Ctx);

    // 5) output projection -> fp32
    {
        const int blocks = (MROWS / 16) * (DMODEL / 128) / 4;
        gemm_bias_kernel<<<blocks, 128, 0, stream>>>(Ctx, Wto, bo, out, nullptr,
                                                     MROWS, DMODEL, DMODEL, 0);
    }
}
